// GCNTransformerLayer_36756330119354
// MI455X (gfx1250) — matrix operand order repaired and checked by reading, not compiled, not run on hardware
//
#include <hip/hip_runtime.h>
#include <hip/hip_bf16.h>
#include <math.h>

// Problem constants (match reference)
constexpr int kN  = 1024;
constexpr int kB  = 8;
constexpr int kD  = 768;
constexpr int kH  = 8;
constexpr int kDH = 96;      // kD / kH
constexpr int kE  = 16384;
constexpr int kRows = kN * kB;   // 8192 token rows, row index = n*B + b

typedef __attribute__((ext_vector_type(16))) __bf16 v16bf;
typedef __attribute__((ext_vector_type(8)))  float  v8f;
typedef __attribute__((ext_vector_type(4)))  unsigned int u32x4;
typedef __attribute__((ext_vector_type(8)))  int    i32x8;
typedef __attribute__((ext_vector_type(4)))  int    i32x4;

union Frag16 {
  v16bf v;
  unsigned short u[16];
  u32x4 q[2];
};

static __device__ inline unsigned short f2bf(float f) {
  unsigned u = __float_as_uint(f);
  u += 0x7FFFu + ((u >> 16) & 1u);   // round-to-nearest-even to bf16
  return (unsigned short)(u >> 16);
}

// -----------------------------------------------------------------------------
// TDM: DMA one 2-D bf16 tile (32 K-elems x 64 rows, row stride = K elems) from
// global into LDS, padding 16B after every 64B row so LDS rows are 80B apart.
// D# packing per CDNA5 ISA ch.8 (groups 0/1; groups 2/3 zero for 2-D).
// This toolchain exposes the 6-arg builtin (g0, g1, g2, g3, g_extra, cpol).
// -----------------------------------------------------------------------------
static __device__ inline void tdm_load_tile_2d(unsigned lds_addr,
                                               const unsigned short* gptr,
                                               int strideElems) {
  unsigned long long ga = (unsigned long long)(uintptr_t)gptr;
  u32x4 g0;
  g0[0] = 1u;                                   // count=1, user descriptor
  g0[1] = lds_addr;                             // LDS byte address
  g0[2] = (unsigned)(ga & 0xFFFFFFFFu);         // global_addr[31:0]
  g0[3] = (unsigned)((ga >> 32) & 0x01FFFFFFu)  // global_addr[56:32]
          | 0x80000000u;                        // type=2 ("image") in bits 127:126
  i32x8 g1;
  g1[0] = (int)((1u << 16)      // data_size = 1 -> 2 bytes
              | (1u << 20)      // pad_enable
              | (3u << 22)      // pad_interval code 3 -> every 16 dwords (64B)
              | (3u << 25));    // pad_amount  code 3 -> 4 dwords (16B)
  g1[1] = (int)(32u << 16);     // tensor_dim0 = 32 (bits 79:48, low 16 here)
  g1[2] = (int)(64u << 16);     // tensor_dim0 hi=0 ; tensor_dim1 = 64 (bits 111:80)
  g1[3] = (int)(32u << 16);     // tensor_dim1 hi=0 ; tile_dim0 = 32 (bits 127:112)
  g1[4] = (int)(64u);           // tile_dim1 = 64 ; tile_dim2 = 0 (2-D)
  g1[5] = strideElems;          // tensor_dim0_stride[31:0] (elements)
  g1[6] = 0;                    // stride hi ; tensor_dim1_stride lo (unused, 2-D)
  g1[7] = 0;
  i32x4 z4 = {0, 0, 0, 0};
  i32x8 z8 = {0, 0, 0, 0, 0, 0, 0, 0};
  __builtin_amdgcn_tensor_load_to_lds(g0, g1, z4, z4, z8, 0);
}

// -----------------------------------------------------------------------------
// Generic GEMM: Y[M,Nn] = A[M,K](bf16) @ W[Nn,K](bf16)^T (+ bias[Nn]), Y f32.
// Block: 128 threads (4 wave32). Tile 64(M) x 64(N), K-step 32.
// Tiles staged by the Tensor Data Mover; fragments are two ds_load_b128 each.
// -----------------------------------------------------------------------------
__global__ __launch_bounds__(128)
void gemm_xwT_bf16wmma(const unsigned short* __restrict__ X,
                       const unsigned short* __restrict__ W,
                       const float* __restrict__ bias, float* __restrict__ Y,
                       int M, int Nn, int K) {
  __shared__ unsigned short As[64][40];   // [m][k] bf16, 80B padded rows
  __shared__ unsigned short Bs[64][40];   // [n][k] bf16, 80B padded rows
  const int tid  = threadIdx.x;
  const int wid  = tid >> 5;
  const int lane = tid & 31;
  const int l    = lane & 15;
  const int hi   = lane >> 4;
  const int m0   = blockIdx.y * 64;
  const int j0   = blockIdx.x * 64;

  const unsigned ldsA = (unsigned)(uintptr_t)&As[0][0];
  const unsigned ldsB = (unsigned)(uintptr_t)&Bs[0][0];

  v8f acc[4] = {};

  for (int k0 = 0; k0 < K; k0 += 32) {
    if (tid == 0) {   // wave 0 issues both DMAs (EXEC ignored by TDM)
      tdm_load_tile_2d(ldsA, X + (size_t)m0 * K + k0, K);
      tdm_load_tile_2d(ldsB, W + (size_t)j0 * K + k0, K);
      __builtin_amdgcn_s_wait_tensorcnt(0);
    }
    __syncthreads();

    // A fragment: u[0..7] = As[row][hi*16 .. +7], u[8..15] = As[row][hi*16+8 .. +15]
    Frag16 af;
    {
      const unsigned short* ap = &As[wid * 16 + l][0];
      af.q[0] = *(const u32x4*)(ap + hi * 16);
      af.q[1] = *(const u32x4*)(ap + hi * 16 + 8);
    }

    #pragma unroll
    for (int t = 0; t < 4; ++t) {
      // B fragment: u[e] = Bs[col][hi*16 + e] -> one contiguous 32B run
      Frag16 bfr;
      const unsigned short* bp = &Bs[t * 16 + l][hi * 16];
      bfr.q[0] = ((const u32x4*)bp)[0];
      bfr.q[1] = ((const u32x4*)bp)[1];
      acc[t] = __builtin_amdgcn_wmma_f32_16x16x32_bf16(
          false, af.v, false, bfr.v, (short)0, acc[t], false, false);
    }
    __syncthreads();
  }

  // C/D layout: VGPR r -> row (r + hi*8), col = lane&15
  #pragma unroll
  for (int t = 0; t < 4; ++t)
    #pragma unroll
    for (int r = 0; r < 8; ++r) {
      int m   = m0 + wid * 16 + hi * 8 + r;
      int col = j0 + t * 16 + l;
      float v = acc[t][r];
      if (bias) v += bias[col];
      Y[(size_t)m * Nn + col] = v;
    }
}

// -----------------------------------------------------------------------------
// f32 -> packed bf16 conversion (one-time per operand, out of the GEMM loop)
// -----------------------------------------------------------------------------
__global__ void f32_to_bf16_kernel(const float* __restrict__ src,
                                   unsigned short* __restrict__ dst, size_t n) {
  for (size_t i = (size_t)blockIdx.x * 256 + threadIdx.x; i < n;
       i += (size_t)gridDim.x * 256)
    dst[i] = f2bf(src[i]);
}

// -----------------------------------------------------------------------------
// GCN helpers
// -----------------------------------------------------------------------------
__global__ void deg_init_kernel(float* deg) {
  int i = blockIdx.x * 256 + threadIdx.x;
  if (i < kB * kN) deg[i] = 1.0f;   // self-loop contributes 1
}

__global__ void deg_accum_kernel(const int* __restrict__ ei, float* deg) {
  int i = blockIdx.x * 256 + threadIdx.x;
  if (i >= kB * kE) return;
  int b = i / kE, e = i % kE;
  int dst = ei[(size_t)b * 2 * kE + kE + e];
  atomicAdd(&deg[b * kN + dst], 1.0f);
}

__global__ void deg_to_dinv_kernel(float* deg) {
  int i = blockIdx.x * 256 + threadIdx.x;
  if (i < kB * kN) deg[i] = rsqrtf(deg[i]);
}

// agg[(n,b),o] = dinv[b,n]^2 * h[(n,b),o]   (self loop)
__global__ void gcn_self_kernel(const float* __restrict__ h,
                                const float* __restrict__ dinv,
                                float* __restrict__ agg) {
  size_t total = (size_t)kRows * kD;
  for (size_t idx = (size_t)blockIdx.x * 256 + threadIdx.x; idx < total;
       idx += (size_t)gridDim.x * 256) {
    int r = (int)(idx / kD);
    int n = r / kB, b = r % kB;
    float di = dinv[b * kN + n];
    agg[idx] = di * di * h[idx];
  }
}

// one block (256 threads) per (b,e); each thread does 3 of 768 channels
__global__ void gcn_scatter_kernel(const float* __restrict__ h,
                                   const float* __restrict__ dinv,
                                   const int* __restrict__ ei,
                                   float* __restrict__ agg) {
  int be = blockIdx.x;
  int b = be / kE, e = be % kE;
  int src = ei[(size_t)b * 2 * kE + e];
  int dst = ei[(size_t)b * 2 * kE + kE + e];
  float norm = dinv[b * kN + src] * dinv[b * kN + dst];
  const float* hs = h   + ((size_t)src * kB + b) * kD;
  float*       ad = agg + ((size_t)dst * kB + b) * kD;
  #pragma unroll
  for (int i = 0; i < 3; ++i) {
    int o = threadIdx.x + i * 256;
    atomicAdd(&ad[o], norm * hs[o]);
  }
}

// -----------------------------------------------------------------------------
// Attention: one wave32 per query, online softmax over 1024 keys.
// Writes bf16 context directly (feeds the out_proj WMMA GEMM).
// -----------------------------------------------------------------------------
__global__ __launch_bounds__(256)
void attn_kernel(const float* __restrict__ qkv, unsigned short* __restrict__ ctxbf) {
  const int wid  = threadIdx.x >> 5;
  const int lane = threadIdx.x & 31;
  int qid = blockIdx.x * 8 + wid;            // over B*H*N
  if (qid >= kB * kH * kN) return;
  int b = qid / (kH * kN);
  int rem = qid % (kH * kN);
  int h = rem / kN;
  int n = rem % kN;

  const float scale = 0.1020620726f;          // 1/sqrt(96)
  const float* qp = qkv + ((size_t)n * kB + b) * (3 * kD) + h * kDH;
  float q0 = qp[lane], q1 = qp[lane + 32], q2 = qp[lane + 64];

  float m = -1e30f, lsum = 0.0f, a0 = 0.0f, a1 = 0.0f, a2 = 0.0f;
  for (int kk = 0; kk < kN; ++kk) {
    const float* kp = qkv + ((size_t)kk * kB + b) * (3 * kD) + kD + h * kDH;
    float s = q0 * kp[lane] + q1 * kp[lane + 32] + q2 * kp[lane + 64];
    #pragma unroll
    for (int o = 16; o >= 1; o >>= 1) s += __shfl_xor(s, o, 32);
    s *= scale;
    float mn   = fmaxf(m, s);
    float corr = __expf(m - mn);
    float p    = __expf(s - mn);
    const float* vp = kp + kD;
    lsum = lsum * corr + p;
    a0 = a0 * corr + p * vp[lane];
    a1 = a1 * corr + p * vp[lane + 32];
    a2 = a2 * corr + p * vp[lane + 64];
    m = mn;
  }
  float inv = 1.0f / lsum;
  unsigned short* cp = ctxbf + ((size_t)n * kB + b) * kD + h * kDH;
  cp[lane]      = f2bf(a0 * inv);
  cp[lane + 32] = f2bf(a1 * inv);
  cp[lane + 64] = f2bf(a2 * inv);
}

// combined f32 (for final fuse) and bf16 (for gate GEMM): [gcn_out | attn_out]
__global__ void combine_kernel(const float* __restrict__ agg,
                               const float* __restrict__ gcn_b,
                               const float* __restrict__ attn,
                               float* __restrict__ combined,
                               unsigned short* __restrict__ combined_bf) {
  size_t total = (size_t)kRows * kD;
  for (size_t idx = (size_t)blockIdx.x * 256 + threadIdx.x; idx < total;
       idx += (size_t)gridDim.x * 256) {
    int r = (int)(idx / kD), o = (int)(idx % kD);
    float g = agg[idx] + gcn_b[o];
    float a = attn[idx];
    combined[(size_t)r * 2 * kD + o]          = g;
    combined[(size_t)r * 2 * kD + kD + o]     = a;
    combined_bf[(size_t)r * 2 * kD + o]       = f2bf(g);
    combined_bf[(size_t)r * 2 * kD + kD + o]  = f2bf(a);
  }
}

// sigmoid gate, fuse, residual, layernorm — one block (256 thr) per row
__global__ __launch_bounds__(256)
void fuse_ln_kernel(const float* __restrict__ gate_lin,
                    const float* __restrict__ combined,
                    const float* __restrict__ x,
                    const float* __restrict__ ln_g,
                    const float* __restrict__ ln_b,
                    float* __restrict__ out) {
  int r = blockIdx.x;
  const float* gl = gate_lin + (size_t)r * kD;
  const float* cb = combined + (size_t)r * 2 * kD;
  const float* xr = x + (size_t)r * kD;

  float f[3], sum = 0.0f, sq = 0.0f;
  #pragma unroll
  for (int i = 0; i < 3; ++i) {
    int o = threadIdx.x + i * 256;
    float g  = 1.0f / (1.0f + __expf(-gl[o]));
    float fv = g * cb[o] + (1.0f - g) * cb[kD + o] + xr[o];
    f[i] = fv; sum += fv; sq += fv * fv;
  }
  __shared__ float rs[256], rq[256];
  rs[threadIdx.x] = sum; rq[threadIdx.x] = sq;
  __syncthreads();
  for (int s = 128; s > 0; s >>= 1) {
    if (threadIdx.x < s) {
      rs[threadIdx.x] += rs[threadIdx.x + s];
      rq[threadIdx.x] += rq[threadIdx.x + s];
    }
    __syncthreads();
  }
  float mean = rs[0] * (1.0f / kD);
  float var  = rq[0] * (1.0f / kD) - mean * mean;
  float rstd = rsqrtf(var + 1e-5f);
  #pragma unroll
  for (int i = 0; i < 3; ++i) {
    int o = threadIdx.x + i * 256;
    out[(size_t)r * kD + o] = (f[i] - mean) * rstd * ln_g[o] + ln_b[o];
  }
}

// -----------------------------------------------------------------------------
extern "C" void kernel_launch(void* const* d_in, const int* in_sizes, int n_in,
                              void* d_out, int out_size, void* d_ws, size_t ws_size,
                              hipStream_t stream) {
  (void)in_sizes; (void)n_in; (void)out_size; (void)ws_size;
  const float* x      = (const float*)d_in[0];
  const int*   ei     = (const int*)  d_in[1];
  const float* gcn_W  = (const float*)d_in[2];
  const float* gcn_b  = (const float*)d_in[3];
  const float* in_w   = (const float*)d_in[4];
  const float* in_b   = (const float*)d_in[5];
  const float* out_w  = (const float*)d_in[6];
  const float* out_b  = (const float*)d_in[7];
  const float* gate_w = (const float*)d_in[8];
  const float* gate_b = (const float*)d_in[9];
  const float* ln_g   = (const float*)d_in[10];
  const float* ln_b   = (const float*)d_in[11];
  float* out = (float*)d_out;

  // Workspace layout (SZ = 25,165,824 B):
  //   [0   ,  SZ)  h (f32)            -> reused as gate_lin after scatter
  //   [SZ  , 2SZ)  agg (f32)
  //   [2SZ , 5SZ)  qkv (f32)          -> attn_out [2SZ,3SZ) + combined f32 [3SZ,5SZ)
  //   [5SZ , 5.5SZ) xbf  (bf16 of x)
  //   [5.5SZ, 6SZ)  ctxbf (bf16 ctx)
  //   [6SZ , 7SZ)  combined_bf (bf16)
  //   [7SZ , +4MB) wbf (bf16 weight staging, max 3.54MB)
  //   then deg (32KB)
  char* ws = (char*)d_ws;
  const size_t SZ = (size_t)kRows * kD * sizeof(float);
  float*          h        = (float*)(ws);
  float*          agg      = (float*)(ws + SZ);
  float*          qkv      = (float*)(ws + 2 * SZ);
  float*          attn     = (float*)(ws + 2 * SZ);          // reuse qkv
  float*          combined = (float*)(ws + 3 * SZ);
  unsigned short* xbf      = (unsigned short*)(ws + 5 * SZ);
  unsigned short* ctxbf    = (unsigned short*)(ws + 5 * SZ + SZ / 2);
  unsigned short* cmbbf    = (unsigned short*)(ws + 6 * SZ);
  unsigned short* wbf      = (unsigned short*)(ws + 7 * SZ);
  float*          deg      = (float*)(ws + 7 * SZ + (size_t)4 * 1024 * 1024);
  float*          gate_lin = h;                               // reuse (h dead after scatter)

  // 0. pre-convert x to bf16
  f32_to_bf16_kernel<<<4096, 256, 0, stream>>>(x, xbf, (size_t)kRows * kD);

  // 1. GCN linear: h = x @ gcn_W^T (bias applied post-aggregation)
  f32_to_bf16_kernel<<<1152, 256, 0, stream>>>(gcn_W, wbf, (size_t)kD * kD);
  { dim3 g(kD / 64, kRows / 64);
    gemm_xwT_bf16wmma<<<g, 128, 0, stream>>>(xbf, wbf, nullptr, h, kRows, kD, kD); }

  // 2. degrees -> dinv
  deg_init_kernel<<<(kB * kN + 255) / 256, 256, 0, stream>>>(deg);
  deg_accum_kernel<<<(kB * kE + 255) / 256, 256, 0, stream>>>(ei, deg);
  deg_to_dinv_kernel<<<(kB * kN + 255) / 256, 256, 0, stream>>>(deg);

  // 3. GCN aggregation: self loops + edge scatter (L2-resident atomics)
  gcn_self_kernel<<<4096, 256, 0, stream>>>(h, deg, agg);
  gcn_scatter_kernel<<<kB * kE, 256, 0, stream>>>(h, deg, ei, agg);

  // 4. QKV projection
  f32_to_bf16_kernel<<<3456, 256, 0, stream>>>(in_w, wbf, (size_t)3 * kD * kD);
  { dim3 g((3 * kD) / 64, kRows / 64);
    gemm_xwT_bf16wmma<<<g, 128, 0, stream>>>(xbf, wbf, in_b, qkv, kRows, 3 * kD, kD); }

  // 5. Attention -> ctx (bf16)
  attn_kernel<<<(kB * kH * kN) / 8, 256, 0, stream>>>(qkv, ctxbf);

  // 6. Output projection -> attn_out (overwrites dead qkv region)
  f32_to_bf16_kernel<<<1152, 256, 0, stream>>>(out_w, wbf, (size_t)kD * kD);
  { dim3 g(kD / 64, kRows / 64);
    gemm_xwT_bf16wmma<<<g, 128, 0, stream>>>(ctxbf, wbf, out_b, attn, kRows, kD, kD); }

  // 7. combined = [gcn_out | attn_out] (f32 + bf16)
  combine_kernel<<<4096, 256, 0, stream>>>(agg, gcn_b, attn, combined, cmbbf);

  // 8. gate logits (K = 2D)
  f32_to_bf16_kernel<<<2304, 256, 0, stream>>>(gate_w, wbf, (size_t)kD * 2 * kD);
  { dim3 g(kD / 64, kRows / 64);
    gemm_xwT_bf16wmma<<<g, 128, 0, stream>>>(cmbbf, wbf, gate_b, gate_lin,
                                             kRows, kD, 2 * kD); }

  // 9. gate, fuse, residual, layernorm
  fuse_ln_kernel<<<kRows, 256, 0, stream>>>(gate_lin, combined, x, ln_g, ln_b, out);
}
